// SPInterAttModule_15178414424521
// MI455X (gfx1250) — matrix operand order repaired and checked by reading, not compiled, not run on hardware
//
#include <hip/hip_runtime.h>

// ---------------------------------------------------------------------------
// SPInterAttModule for MI455X (gfx1250), bf16 WMMA pipeline + TDM async loads.
// B=8, C=256, H=W=64 (N=4096), M=256, NH=8, dh=32, QK=256.
// ---------------------------------------------------------------------------

#define B_   8
#define C_   256
#define N_   4096
#define M_   256
#define NH_  8
#define DH_  32
#define QK_  256
#define SCALE_ 0.17677669529663687f   // 1/sqrt(32)
#define EPS_LN_ 1e-6f
#define EPS_SP_ 1e-16f

typedef unsigned short u16;
typedef __attribute__((ext_vector_type(16))) __bf16 v16bf;
typedef __attribute__((ext_vector_type(8)))  float  v8f;
typedef __attribute__((ext_vector_type(4)))  unsigned int v4u_;
typedef __attribute__((ext_vector_type(4)))  int v4i_;
typedef __attribute__((ext_vector_type(8)))  int v8i_;

// ---------------- small helpers ----------------
static __device__ inline u16 f2bf(float f) {
    union { float f; unsigned u; } c; c.f = f;
    unsigned u = c.u;
    unsigned r = u + 0x7FFFu + ((u >> 16) & 1u);
    if ((u & 0x7F800000u) == 0x7F800000u) r = u;   // inf / nan passthrough
    return (u16)(r >> 16);
}
static __device__ inline __bf16 bfbits(u16 b) {
    union { u16 u; __bf16 h; } c; c.u = b; return c.h;
}
static __device__ inline v8f zero8() {
    v8f z = {0.f,0.f,0.f,0.f,0.f,0.f,0.f,0.f}; return z;
}

// ---------------- TDM: async 2-D tensor tile load, global -> LDS ------------
// Descriptor per CDNA5 ISA ch.8: group0 = {count/lds_addr/global_addr/type},
// group1 = {mask, data_size=2B, tensor dims, tile dims, dim0 stride}.
// tensor_d*/tile_d*/stride0 are in elements (bf16).
static __device__ inline void tdm_load_2d_bf16(const void* gptr, unsigned ldsOff,
                                               unsigned tensor_d0, unsigned tensor_d1,
                                               unsigned tile_d0, unsigned tile_d1,
                                               unsigned stride0) {
    unsigned long long ga = (unsigned long long)gptr;
    v4u_ g0;
    g0[0] = 1u;                                             // count=1 (valid), user mode
    g0[1] = ldsOff;                                         // lds_addr (bytes)
    g0[2] = (unsigned)ga;                                   // global_addr[31:0]
    g0[3] = (unsigned)((ga >> 32) & 0x1FFFFFFu) | (2u << 30); // addr[56:32] | type=2
    v8i_ g1;
    g1[0] = (int)(1u << 16);                                // wg_mask=0, data_size=1 (2B)
    g1[1] = (int)((tensor_d0 & 0xFFFFu) << 16);             // tensor_dim0[15:0]
    g1[2] = (int)(((tensor_d0 >> 16) & 0xFFFFu) |
                  ((tensor_d1 & 0xFFFFu) << 16));           // tdim0 hi | tdim1 lo
    g1[3] = (int)(((tensor_d1 >> 16) & 0xFFFFu) |
                  ((tile_d0 & 0xFFFFu) << 16));             // tdim1 hi | tile_dim0
    g1[4] = (int)(tile_d1 & 0xFFFFu);                       // tile_dim1 | tile_dim2=0
    g1[5] = (int)stride0;                                   // tensor_dim0_stride lo32
    g1[6] = 0;                                              // stride hi | dim1_stride lo
    g1[7] = 0;
    v4i_ z4 = {0, 0, 0, 0};
#if defined(__clang_major__) && (__clang_major__ >= 23)
    v8i_ z8 = {0, 0, 0, 0, 0, 0, 0, 0};
    __builtin_amdgcn_tensor_load_to_lds(g0, g1, z4, z4, z8, 0);
#else
    __builtin_amdgcn_tensor_load_to_lds(g0, g1, z4, z4, 0);
#endif
}

// A fragment: 16x32 bf16 (MxK), row-major in LDS with leading dim lda (elems).
// ISA 7.12.2: lanes 0-15 -> M=lane, k in {0..7,16..23}; lanes 16-31 -> k {8..15,24..31}.
static __device__ inline v16bf load_a_frag(const u16* A, int lda) {
    int lane = threadIdx.x & 31;
    int m  = lane & 15;
    int kb = (lane < 16) ? 0 : 8;
    v16bf a;
#pragma unroll
    for (int i = 0; i < 8; ++i) a[i]     = bfbits(A[m * lda + kb + i]);
#pragma unroll
    for (int i = 0; i < 8; ++i) a[8 + i] = bfbits(A[m * lda + kb + 16 + i]);
    return a;
}

// B fragment: 32x16 bf16 (KxN), row-major in LDS; lane L holds column n=L%16, k in {kb..kb+7, kb+16..kb+23}, kb=0/8.
static __device__ inline v16bf load_b_frag(const u16* Bm, int ldb) {
    int lane = threadIdx.x & 31;
    int n  = lane & 15;
    int kb = (lane < 16) ? 0 : 8;
    v16bf b;
#pragma unroll
    for (int i = 0; i < 8; ++i) b[i]     = bfbits(Bm[(kb + i) * ldb + n]);
#pragma unroll
    for (int i = 0; i < 8; ++i) b[8 + i] = bfbits(Bm[(kb + 16 + i) * ldb + n]);
    return b;
}

// ---------------- kernel 0: LayerNorm2d -> bf16 xn (B*N, C) ----------------
__global__ void ln_kernel(const float* __restrict__ x,
                          const float* __restrict__ gamma,
                          const float* __restrict__ beta,
                          u16* __restrict__ xn) {
    int p = blockIdx.x * 8 + (threadIdx.x >> 5);   // pixel index in [0, B*N)
    int b = p >> 12;                               // /N_
    int n = p & (N_ - 1);
    int lane = threadIdx.x & 31;

    float v[8]; float s = 0.f, ss = 0.f;
#pragma unroll
    for (int i = 0; i < 8; ++i) {
        int c = lane * 8 + i;
        float t = x[((long)b * C_ + c) * N_ + n];
        v[i] = t; s += t; ss += t * t;
    }
#pragma unroll
    for (int o = 16; o > 0; o >>= 1) {
        s  += __shfl_xor(s,  o, 32);
        ss += __shfl_xor(ss, o, 32);
    }
    float mu  = s * (1.f / C_);
    float var = ss * (1.f / C_) - mu * mu;
    float r = rsqrtf(var + EPS_LN_);
#pragma unroll
    for (int i = 0; i < 8; ++i) {
        int c = lane * 8 + i;
        xn[((long)(b * N_ + n)) * C_ + c] = f2bf((v[i] - mu) * r * gamma[c] + beta[c]);
    }
}

// ---------------- kernel 1: mask -> bf16 + row sums ----------------
__global__ void mask_prep_kernel(const float* __restrict__ mask,
                                 u16* __restrict__ mask_bf,
                                 float* __restrict__ msum) {
    __shared__ float red[8];
    int row = blockIdx.x, b = blockIdx.y;
    long base = ((long)b * M_ + row) * N_;
    int tid = threadIdx.x;
    float s = 0.f;
#pragma unroll
    for (int i = 0; i < 16; ++i) {
        int idx = i * 256 + tid;
        float t = mask[base + idx];
        s += t;
        mask_bf[base + idx] = f2bf(t);
    }
#pragma unroll
    for (int o = 16; o > 0; o >>= 1) s += __shfl_xor(s, o, 32);
    if ((tid & 31) == 0) red[tid >> 5] = s;
    __syncthreads();
    if (tid == 0) {
        float tot = 0.f;
#pragma unroll
        for (int wv = 0; wv < 8; ++wv) tot += red[wv];
        msum[(long)b * M_ + row] = tot;
    }
}

// ---------------- kernel 2: weight transpose + cast: (O,C) f32 -> (C,O) bf16
__global__ void wtrans_kernel(const float* __restrict__ src, u16* __restrict__ dst) {
    int idx = blockIdx.x * 256 + threadIdx.x;    // over C_*QK_
    int c = idx >> 8;         // / QK_
    int o = idx & 255;
    dst[(long)c * QK_ + o] = f2bf(src[(long)o * C_ + c]);
}

// ---------------- kernel 3: generic bf16 GEMM, 64x64 tile, 256 threads ------
// C(MxN) = A(MxK) * B(KxN), all row-major bf16, f32 accum, bf16 out.
// Optional per-row divisor: out *= 1/(rowDiv[row]+eps)  (superpixel pooling)
__global__ void gemm_bf16_kernel(const u16* __restrict__ A, long aStride, int lda,
                                 const u16* __restrict__ Bm, long bStride, int ldb,
                                 u16* __restrict__ Co, long cStride, int ldc,
                                 const float* __restrict__ rowDiv,
                                 int Mdim, int Ndim, int Kdim) {
    extern __shared__ __align__(16) char smem[];
    u16* sA = (u16*)smem;          // 64 x 32
    u16* sB = sA + 64 * 32;        // 32 x 64
    (void)Ndim;

    int bz = blockIdx.z;
    const u16* Ab = A  + (long)bz * aStride;
    const u16* Bb = Bm + (long)bz * bStride;
    u16*       Cb = Co + (long)bz * cStride;
    int mtile = blockIdx.y * 64, ntile = blockIdx.x * 64;
    int tid = threadIdx.x, w = tid >> 5, lane = tid & 31;

    v8f acc0 = zero8(), acc1 = zero8();
    int t0 = w, t1 = w + 8;        // each wave owns 2 of the 16 output tiles
    int tm0 = t0 >> 2, tn0 = t0 & 3;
    int tm1 = t1 >> 2, tn1 = t1 & 3;

    for (int kk = 0; kk < Kdim; kk += 32) {
#pragma unroll
        for (int i = 0; i < 8; ++i) {
            int idx = tid * 8 + i;
            int r = idx >> 5, c = idx & 31;
            sA[idx] = Ab[(long)(mtile + r) * lda + kk + c];
        }
#pragma unroll
        for (int i = 0; i < 8; ++i) {
            int idx = tid * 8 + i;
            int r = idx >> 6, c = idx & 63;
            sB[idx] = Bb[(long)(kk + r) * ldb + ntile + c];
        }
        if (kk + 32 < Kdim)
            __builtin_prefetch(&Ab[(long)(mtile + (tid >> 2)) * lda + kk + 32], 0, 1);
        __syncthreads();
        {
            v16bf af = load_a_frag(sA + tm0 * 16 * 32, 32);
            v16bf bf = load_b_frag(sB + tn0 * 16, 64);
            acc0 = __builtin_amdgcn_wmma_f32_16x16x32_bf16(false, af, false, bf,
                                                           (short)0, acc0, false, false);
        }
        {
            v16bf af = load_a_frag(sA + tm1 * 16 * 32, 32);
            v16bf bf = load_b_frag(sB + tn1 * 16, 64);
            acc1 = __builtin_amdgcn_wmma_f32_16x16x32_bf16(false, af, false, bf,
                                                           (short)0, acc1, false, false);
        }
        __syncthreads();
    }

    int col = lane & 15, rb = (lane < 16) ? 0 : 8;
#pragma unroll
    for (int i = 0; i < 8; ++i) {
        int r0 = mtile + tm0 * 16 + rb + i;
        int r1 = mtile + tm1 * 16 + rb + i;
        float f0 = 1.f, f1 = 1.f;
        if (rowDiv) {
            f0 = 1.f / (rowDiv[(long)bz * Mdim + r0] + EPS_SP_);
            f1 = 1.f / (rowDiv[(long)bz * Mdim + r1] + EPS_SP_);
        }
        Cb[(long)r0 * ldc + ntile + tn0 * 16 + col] = f2bf(acc0[i] * f0);
        Cb[(long)r1 * ldc + ntile + tn1 * 16 + col] = f2bf(acc1[i] * f1);
    }
}

// ---------------- kernel 4: stage-1 flash (pixels -> superpixels) -----------
// Per (b, h, m-block of 64): loop n chunks of 64, online softmax over n,
// accumulate s_out(32 x 64) in WMMA fragments. K tile comes in via TDM;
// V tile is staged transposed (d x n) so V^T A-fragments vectorize.
__global__ void stage1_kernel(const u16* __restrict__ Kb,
                              const u16* __restrict__ Vb,
                              const u16* __restrict__ sth,
                              u16* __restrict__ sout) {
    extern __shared__ __align__(16) char smem[];
    u16*  sK  = (u16*)smem;           // 64 x 32  (n x d)   <- TDM
    u16*  sVt = sK + 64 * 32;         // 32 x 64  (d x n)   transposed stage
    u16*  sS  = sVt + 32 * 64;        // 32 x 64  (d x m)
    u16*  sP  = sS + 32 * 64;         // 64 x 64  (n x m) bf16
    float* sL   = (float*)(sP + 64 * 64);  // 64 x 64 logits
    float* cmax = sL + 64 * 64;
    float* csum = cmax + 64;
    float* cfac = csum + 64;

    int b = blockIdx.z, h = blockIdx.y, mb0 = blockIdx.x * 64;
    int tid = threadIdx.x, w = tid >> 5, lane = tid & 31;
    int sub_d = w >> 2, sub_m = w & 3;
    int col = lane & 15, rb = (lane < 16) ? 0 : 8;
    unsigned ldsK = (unsigned)((const char*)sK - (const char*)smem) +
                    __builtin_amdgcn_groupstaticsize();

    for (int i = tid; i < 32 * 64; i += 256) {
        int d = i >> 6, m = i & 63;
        sS[i] = sth[((long)(b * M_ + mb0 + m)) * C_ + h * DH_ + d];
    }
    if (tid < 64) { cmax[tid] = -3.0e38f; csum[tid] = 0.f; }
    v8f acc = zero8();
    __syncthreads();

    for (int n0 = 0; n0 < N_; n0 += 64) {
        // async DMA: 64(n) x 32(d) bf16 tile of K into sK
        if (tid == 0) {
            const u16* kg = Kb + ((long)(b * N_ + n0)) * QK_ + h * DH_;
            tdm_load_2d_bf16(kg, ldsK, 32, 64, 32, 64, QK_);
        }
        // manual transposed stage of V (overlaps with TDM)
        for (int i = tid; i < 2048; i += 256) {
            int r = i >> 5, c = i & 31;   // coalesced global read (r=n, c=d)
            sVt[c * 64 + r] = Vb[((long)(b * N_ + n0 + r)) * QK_ + h * DH_ + c];
        }
        if (tid == 0) __builtin_amdgcn_s_wait_tensorcnt(0);
        __syncthreads();

        // logits tiles: 16 tiles of 16x16 (n x m), K=32(=dh), 2 per wave
        for (int t = w; t < 16; t += 8) {
            int tn = t >> 2, tm = t & 3;
            v16bf af = load_a_frag(sK + tn * 16 * 32, 32);
            v16bf bf = load_b_frag(sS + tm * 16, 64);
            v8f c = __builtin_amdgcn_wmma_f32_16x16x32_bf16(false, af, false, bf,
                                                            (short)0, zero8(), false, false);
#pragma unroll
            for (int i = 0; i < 8; ++i)
                sL[(tn * 16 + rb + i) * 64 + tm * 16 + col] = c[i] * SCALE_;
        }
        __syncthreads();

        // online softmax stats per m column
        if (tid < 64) {
            int m = tid;
            float om = cmax[m], cm = -3.0e38f;
            for (int r = 0; r < 64; ++r) cm = fmaxf(cm, sL[r * 64 + m]);
            float nm = fmaxf(om, cm);
            float f  = __expf(om - nm);
            float s  = 0.f;
            for (int r = 0; r < 64; ++r) s += __expf(sL[r * 64 + m] - nm);
            csum[m] = csum[m] * f + s;
            cmax[m] = nm;
            cfac[m] = f;
        }
        __syncthreads();

        // P = exp(L - max) as bf16; rescale accumulators per column
        for (int i = tid; i < 4096; i += 256) {
            int m = i & 63;
            sP[i] = f2bf(__expf(sL[i] - cmax[m]));
        }
        {
            float f = cfac[sub_m * 16 + col];
#pragma unroll
            for (int i = 0; i < 8; ++i) acc[i] *= f;
        }
        __syncthreads();

        // acc(16d x 16m) += V^T chunk (16d x 32n) * P chunk (32n x 16m), x2
#pragma unroll
        for (int kc = 0; kc < 2; ++kc) {
            v16bf af = load_a_frag(sVt + sub_d * 16 * 64 + kc * 32, 64);
            v16bf bfr = load_b_frag(sP + kc * 32 * 64 + sub_m * 16, 64);
            acc = __builtin_amdgcn_wmma_f32_16x16x32_bf16(false, af, false, bfr,
                                                          (short)0, acc, false, false);
        }
        __syncthreads();
    }

    // epilogue: divide by softmax denominator, store s_out[b][h][d][m] bf16
    {
        int mcol = sub_m * 16 + col;
        float inv = 1.0f / csum[mcol];
#pragma unroll
        for (int i = 0; i < 8; ++i) {
            int d = sub_d * 16 + rb + i;
            sout[(((long)(b * NH_ + h)) * DH_ + d) * M_ + mb0 + mcol] = f2bf(acc[i] * inv);
        }
    }
}

// ---------------- kernel 5: stage-2 fused (superpixels -> pixels) -----------
// Per (b, h, n-block of 64): full 256-row softmax column in LDS, then
// out(32 x 64) = s_out(32 x 256) @ P(256 x 64). sth tile comes in via TDM.
__global__ void stage2_kernel(const u16* __restrict__ Qb,
                              const u16* __restrict__ sth,
                              const u16* __restrict__ soutb,
                              float* __restrict__ out) {
    extern __shared__ __align__(16) char smem[];
    u16*  sSth = (u16*)smem;              // 256 x 32 (m x d)  <- TDM
    u16*  sQ   = sSth + 256 * 32;         // 32 x 64  (d x n)
    u16*  sO   = sQ + 32 * 64;            // 32 x 256 (d x m)
    u16*  sP   = sO + 32 * 256;           // 256 x 64 (m x n) bf16
    float* sL  = (float*)(sP + 256 * 64); // 256 x 64 logits
    float* nmx = sL + 256 * 64;
    float* nsm = nmx + 64;

    int b = blockIdx.z, h = blockIdx.y, nb = blockIdx.x * 64;
    int tid = threadIdx.x, w = tid >> 5, lane = tid & 31;
    int col = lane & 15, rb = (lane < 16) ? 0 : 8;

    // async DMA: 256(m) x 32(d) tile of sth, row stride C
    if (tid == 0) {
        const u16* sg = sth + ((long)(b * M_)) * C_ + h * DH_;
        unsigned ldsS = (unsigned)((const char*)sSth - (const char*)smem) +
                        __builtin_amdgcn_groupstaticsize();
        tdm_load_2d_bf16(sg, ldsS, 32, 256, 32, 256, C_);
    }
    for (int i = tid; i < 32 * 64; i += 256) {
        int d = i >> 6, n = i & 63;
        sQ[i] = Qb[((long)(b * N_ + nb + n)) * QK_ + h * DH_ + d];
    }
    for (int i = tid; i < 32 * 256; i += 256) {
        int d = i >> 8, m = i & 255;
        sO[i] = soutb[(((long)(b * NH_ + h)) * DH_ + d) * M_ + m];
    }
    if (tid == 0) __builtin_amdgcn_s_wait_tensorcnt(0);
    __syncthreads();

    // logits: 64 tiles of 16x16 (m x n), 8 per wave, K=32(=dh)
    for (int t = w; t < 64; t += 8) {
        int tm = t >> 2, tn = t & 3;
        v16bf af = load_a_frag(sSth + tm * 16 * 32, 32);
        v16bf bf = load_b_frag(sQ + tn * 16, 64);
        v8f c = __builtin_amdgcn_wmma_f32_16x16x32_bf16(false, af, false, bf,
                                                        (short)0, zero8(), false, false);
#pragma unroll
        for (int i = 0; i < 8; ++i)
            sL[(tm * 16 + rb + i) * 64 + tn * 16 + col] = c[i] * SCALE_;
    }
    __syncthreads();

    // full-column softmax over m (256 rows) per n column
    if (tid < 64) {
        float mx = -3.0e38f;
        for (int m = 0; m < 256; ++m) mx = fmaxf(mx, sL[m * 64 + tid]);
        float s = 0.f;
        for (int m = 0; m < 256; ++m) s += __expf(sL[m * 64 + tid] - mx);
        nmx[tid] = mx;
        nsm[tid] = 1.0f / s;
    }
    __syncthreads();
    for (int i = tid; i < 256 * 64; i += 256) {
        int n = i & 63;
        sP[i] = f2bf(__expf(sL[i] - nmx[n]) * nsm[n]);
    }
    __syncthreads();

    // out tile (16d x 16n) per wave, K = 256 over m
    int sub_d = w >> 2, sub_n = w & 3;
    v8f acc = zero8();
    for (int kk = 0; kk < 256; kk += 32) {
        v16bf af = load_a_frag(sO + sub_d * 16 * 256 + kk, 256);
        v16bf bf = load_b_frag(sP + kk * 64 + sub_n * 16, 64);
        acc = __builtin_amdgcn_wmma_f32_16x16x32_bf16(false, af, false, bf,
                                                      (short)0, acc, false, false);
    }
#pragma unroll
    for (int i = 0; i < 8; ++i) {
        int d  = sub_d * 16 + rb + i;
        int ch = h * DH_ + d;
        int ng = nb + sub_n * 16 + col;
        out[((long)b * C_ + ch) * N_ + ng] = acc[i];
    }
}

// ---------------------------------------------------------------------------
extern "C" void kernel_launch(void* const* d_in, const int* in_sizes, int n_in,
                              void* d_out, int out_size, void* d_ws, size_t ws_size,
                              hipStream_t stream) {
    (void)in_sizes; (void)n_in; (void)out_size; (void)ws_size;
    const float* x     = (const float*)d_in[0];
    const float* mask  = (const float*)d_in[1];
    const float* Wq    = (const float*)d_in[2];
    const float* Wk    = (const float*)d_in[3];
    const float* Wv    = (const float*)d_in[4];
    const float* gamma = (const float*)d_in[5];
    const float* beta  = (const float*)d_in[6];
    float* out = (float*)d_out;

    char* ws = (char*)d_ws;
    size_t off = 0;
    auto carve = [&](size_t bytes) -> char* {
        char* p = ws + off;
        off += (bytes + 255) & ~(size_t)255;
        return p;
    };
    u16*   xn    = (u16*)carve((size_t)B_ * N_ * C_ * 2);
    u16*   mbf   = (u16*)carve((size_t)B_ * M_ * N_ * 2);
    float* msum  = (float*)carve((size_t)B_ * M_ * 4);
    u16*   wqT   = (u16*)carve((size_t)C_ * QK_ * 2);
    u16*   wkT   = (u16*)carve((size_t)C_ * QK_ * 2);
    u16*   wvT   = (u16*)carve((size_t)C_ * QK_ * 2);
    u16*   Qb    = (u16*)carve((size_t)B_ * N_ * QK_ * 2);
    u16*   Kb    = (u16*)carve((size_t)B_ * N_ * QK_ * 2);
    u16*   Vb    = (u16*)carve((size_t)B_ * N_ * QK_ * 2);
    u16*   sth   = (u16*)carve((size_t)B_ * M_ * C_ * 2);
    u16*   sout  = (u16*)carve((size_t)B_ * NH_ * DH_ * M_ * 2);

    // 0) LayerNorm -> bf16 xn
    ln_kernel<<<(B_ * N_) / 8, 256, 0, stream>>>(x, gamma, beta, xn);
    // 1) mask -> bf16 + row sums
    mask_prep_kernel<<<dim3(M_, B_), 256, 0, stream>>>(mask, mbf, msum);
    // 2) weights -> transposed bf16
    wtrans_kernel<<<(C_ * QK_) / 256, 256, 0, stream>>>(Wq, wqT);
    wtrans_kernel<<<(C_ * QK_) / 256, 256, 0, stream>>>(Wk, wkT);
    wtrans_kernel<<<(C_ * QK_) / 256, 256, 0, stream>>>(Wv, wvT);

    size_t gemmLds = (size_t)(64 * 32 + 32 * 64) * sizeof(u16);
    // 3) Q/K/V projections: (B*N x C) * (C x QK)
    gemm_bf16_kernel<<<dim3(QK_ / 64, (B_ * N_) / 64, 1), 256, gemmLds, stream>>>(
        xn, 0, C_, wqT, 0, QK_, Qb, 0, QK_, nullptr, B_ * N_, QK_, C_);
    gemm_bf16_kernel<<<dim3(QK_ / 64, (B_ * N_) / 64, 1), 256, gemmLds, stream>>>(
        xn, 0, C_, wkT, 0, QK_, Kb, 0, QK_, nullptr, B_ * N_, QK_, C_);
    gemm_bf16_kernel<<<dim3(QK_ / 64, (B_ * N_) / 64, 1), 256, gemmLds, stream>>>(
        xn, 0, C_, wvT, 0, QK_, Vb, 0, QK_, nullptr, B_ * N_, QK_, C_);
    // 4) superpixel tokens: stok = mask @ xn, normalized by row sums -> sth
    gemm_bf16_kernel<<<dim3(C_ / 64, M_ / 64, B_), 256, gemmLds, stream>>>(
        mbf, (long)M_ * N_, N_, xn, (long)N_ * C_, C_,
        sth, (long)M_ * C_, C_, msum, M_, C_, N_);

    // 5) stage 1: flash attention over pixel axis (TDM K tiles)
    size_t lds1 = (size_t)(64 * 32 + 2 * 32 * 64 + 64 * 64) * sizeof(u16) +
                  (size_t)(64 * 64 + 3 * 64) * sizeof(float);
    stage1_kernel<<<dim3(M_ / 64, NH_, B_), 256, lds1, stream>>>(Kb, Vb, sth, sout);

    // 6) stage 2: fused softmax over superpixel axis + output GEMM (TDM sth tile)
    size_t lds2 = (size_t)(256 * 32 + 32 * 64 + 32 * 256 + 256 * 64) * sizeof(u16) +
                  (size_t)(256 * 64 + 2 * 64) * sizeof(float);
    stage2_kernel<<<dim3(N_ / 64, NH_, B_), 256, lds2, stream>>>(Qb, sth, sout, out);
}
